// MHA_56977036149043
// MI455X (gfx1250) — matrix operand order repaired and checked by reading, not compiled, not run on hardware
//
#include <hip/hip_runtime.h>

typedef __attribute__((ext_vector_type(16))) _Float16 v16h;
typedef __attribute__((ext_vector_type(8)))  _Float16 v8h;
typedef __attribute__((ext_vector_type(8)))  float    v8f;
typedef __attribute__((ext_vector_type(4)))  unsigned int u32x4;
typedef __attribute__((ext_vector_type(8)))  int      i32x8;
typedef __attribute__((ext_vector_type(4)))  int      i32x4;

typedef _Float16 h16;

#define BB 8
#define TT 1024
#define CC 1024
#define HH 8
#define DD 128
#define MM (BB * TT)   // 8192

// LDS pitches (halves): payload + 16B pad so consecutive rows start 20 (resp. 68)
// banks apart -> conflict-free 16B ds_load starts.
#define LP32  40       // 32-half rows  (64B + 16B pad)
#define LP128 136      // 128-half rows (256B + 16B pad)

#if defined(__has_builtin)
#  if __has_builtin(__builtin_amdgcn_tensor_load_to_lds) && __has_builtin(__builtin_amdgcn_s_wait_tensorcnt)
#    define USE_TDM 1
#  endif
#endif
#ifndef USE_TDM
#  define USE_TDM 0
#endif

union V16U { v16h v; v8h h[2]; };

// ---------------------------------------------------------------------------
// WMMA helpers
// ---------------------------------------------------------------------------
__device__ __forceinline__ v8f wmma_f16(v16h a, v16h b, v8f c) {
    return __builtin_amdgcn_wmma_f32_16x16x32_f16(
        /*neg_a=*/false, a, /*neg_b=*/false, b,
        /*c_mod=*/(short)0, c, /*reuse_a=*/false, /*reuse_b=*/false);
}

// A-fragment 16x32 (MxK) from row-major [rows][ld] f16.
__device__ __forceinline__ v16h load_a16x32(const h16* __restrict__ base, int ld,
                                            int row0, int k0, int lane) {
    int row = row0 + (lane & 15);
    int kk  = k0 + ((lane >> 4) << 3);
    const h16* p = base + (size_t)row * ld + kk;
    V16U u;
    u.h[0] = *(const v8h*)(p);
    u.h[1] = *(const v8h*)(p + 16);
    return u.v;
}

// B-fragment 32x16 (KxN): element (k,n) at base[(n0+n)*ld + k].
__device__ __forceinline__ v16h load_b32x16(const h16* __restrict__ base, int ld,
                                            int n0, int k0, int lane) {
    int n  = n0 + (lane & 15);
    int kk = k0 + ((lane >> 4) << 3);
    const h16* p = base + (size_t)n * ld + kk;
    V16U u;
    u.h[0] = *(const v8h*)(p);
    u.h[1] = *(const v8h*)(p + 16);
    return u.v;
}

#if USE_TDM
// ---------------------------------------------------------------------------
// TDM descriptors (CDNA5 ISA ch.8). Group1 (dims/strides/tile/pad) is built once
// per kernel; per issue we patch only lds_addr + 57-bit global addr in group0.
// Builtin (clang-23 form): (u32x4 g0, i32x8 g1, i32x4, i32x4, i32x8, i32 cpol)
// ---------------------------------------------------------------------------
__device__ __forceinline__ i32x8 tdm_make_g1(unsigned tensor_w, unsigned tensor_h,
                                             unsigned stride_elems,
                                             unsigned tile_w, unsigned tile_h,
                                             unsigned pad_interval, unsigned pad_amount) {
    i32x8 g1;
    // data_size=1 (2B) | pad_enable | pad_interval | pad_amount
    g1[0] = (int)((1u << 16) | (1u << 20) | (pad_interval << 22) | (pad_amount << 25));
    g1[1] = (int)((tensor_w & 0xFFFFu) << 16);
    g1[2] = (int)(((tensor_w >> 16) & 0xFFFFu) | ((tensor_h & 0xFFFFu) << 16));
    g1[3] = (int)(((tensor_h >> 16) & 0xFFFFu) | ((tile_w & 0xFFFFu) << 16));
    g1[4] = (int)(tile_h & 0xFFFFu);
    g1[5] = (int)stride_elems;
    g1[6] = 0;
    g1[7] = 0;
    return g1;
}

__device__ __forceinline__ void tdm_issue(const i32x8& g1, unsigned lds_addr,
                                          const h16* gptr) {
    unsigned long long ga = (unsigned long long)(uintptr_t)gptr;
    u32x4 g0;
    g0[0] = 1u;                                          // count=1
    g0[1] = lds_addr;
    g0[2] = (unsigned)ga;
    g0[3] = (unsigned)((ga >> 32) & 0x01FFFFFFu) | (2u << 30);  // ga[56:32]|type=2
    i32x4 z4 = {0, 0, 0, 0};
    i32x8 z8 = {0, 0, 0, 0, 0, 0, 0, 0};
    __builtin_amdgcn_tensor_load_to_lds(g0, g1, z4, z4, z8, 0);
}
#endif

// ---------------------------------------------------------------------------
// f32 -> f16 cast
// ---------------------------------------------------------------------------
__global__ void cast_f32_f16_kernel(const float* __restrict__ src,
                                    h16* __restrict__ dst, int n) {
    int i = (blockIdx.x * 256 + threadIdx.x) * 4;
    if (i + 3 < n) {
        float4 v = *(const float4*)(src + i);
        dst[i + 0] = (h16)v.x;
        dst[i + 1] = (h16)v.y;
        dst[i + 2] = (h16)v.z;
        dst[i + 3] = (h16)v.w;
    }
}

// ---------------------------------------------------------------------------
// GEMM: Y[m][n] = sum_k A[m][k] * W[n][k] + bias[n]
// mode 0: out f16 -> [B,H,T,D]; mode 1: out f16 -> [B,H,D,T]; mode 2: f32 [M][N]
// Block tile 128x128, 8 waves (4m x 2n), wave tile 32x64, k-step 32.
// A/B k-slabs double-buffered in LDS, staged by the Tensor Data Mover.
// ---------------------------------------------------------------------------
__global__ void __launch_bounds__(256)
gemm_kernel(const h16* __restrict__ A, const h16* __restrict__ W,
            const float* __restrict__ bias, void* __restrict__ out, int mode) {
    int tid  = threadIdx.x;
    int lane = tid & 31;
    int wave = tid >> 5;
    int wm = wave >> 1;                 // 0..3
    int wn = wave & 1;                  // 0..1
    int m_blk = blockIdx.y * 128;
    int n_blk = blockIdx.x * 128;

    __shared__ __align__(16) h16 sA[2][128][LP32];
    __shared__ __align__(16) h16 sB[2][128][LP32];

    v8f acc[2][4] = {};

#if USE_TDM
    i32x8 g1A = tdm_make_g1(CC, MM, CC, 32, 128, 3, 3);   // pad 4DW per 16DW
    i32x8 g1B = tdm_make_g1(CC, CC, CC, 32, 128, 3, 3);
    auto stage = [&](int buf, int k) {
        if (wave == 0) {
            tdm_issue(g1A, (unsigned)(uintptr_t)&sA[buf][0][0],
                      A + (size_t)m_blk * CC + k);
            tdm_issue(g1B, (unsigned)(uintptr_t)&sB[buf][0][0],
                      W + (size_t)n_blk * CC + k);
        }
    };
#else
    auto stage = [&](int buf, int k) {
#pragma unroll
        for (int i = 0; i < 4; ++i) {
            int c   = tid * 4 + i;          // 0..1023 16B chunks
            int row = (c >> 2) & 127;
            int col = (c & 3) * 8;
            if (c < 512) {
                v8h v = *(const v8h*)(A + (size_t)(m_blk + row) * CC + k + col);
                *(v8h*)&sA[buf][row][col] = v;
            } else {
                v8h v = *(const v8h*)(W + (size_t)(n_blk + row) * CC + k + col);
                *(v8h*)&sB[buf][row][col] = v;
            }
        }
    };
#endif

    stage(0, 0);
#if USE_TDM
    if (wave == 0) __builtin_amdgcn_s_wait_tensorcnt(0);
#endif
    __syncthreads();

    int buf = 0;
    for (int k = 0; k < CC; k += 32) {
        if (k + 32 < CC) stage(buf ^ 1, k + 32);

        v16h a0 = load_a16x32(&sA[buf][0][0], LP32, wm * 32,      0, lane);
        v16h a1 = load_a16x32(&sA[buf][0][0], LP32, wm * 32 + 16, 0, lane);
#pragma unroll
        for (int j = 0; j < 4; ++j) {
            v16h bf = load_b32x16(&sB[buf][0][0], LP32, wn * 64 + 16 * j, 0, lane);
            acc[0][j] = wmma_f16(a0, bf, acc[0][j]);
            acc[1][j] = wmma_f16(a1, bf, acc[1][j]);
        }

#if USE_TDM
        if (wave == 0) __builtin_amdgcn_s_wait_tensorcnt(0);
#endif
        __syncthreads();
        buf ^= 1;
    }

    int halfid = lane >> 4;
    int nl = lane & 15;
#pragma unroll
    for (int i = 0; i < 2; ++i)
#pragma unroll
        for (int j = 0; j < 4; ++j) {
            int col = n_blk + wn * 64 + 16 * j + nl;
            float bv = bias[col];
#pragma unroll
            for (int r = 0; r < 8; ++r) {
                int row = m_blk + wm * 32 + 16 * i + 8 * halfid + r;
                float val = acc[i][j][r] + bv;
                if (mode == 2) {
                    ((float*)out)[(size_t)row * CC + col] = val;
                } else {
                    int bb = row >> 10, t = row & (TT - 1);
                    int hh = col >> 7,  d = col & (DD - 1);
                    h16* oh = (h16*)out;
                    if (mode == 0)
                        oh[(((size_t)(bb * HH + hh)) * TT + t) * DD + d] = (h16)val;
                    else
                        oh[(((size_t)(bb * HH + hh)) * DD + d) * TT + t] = (h16)val;
                }
            }
        }
}

// ---------------------------------------------------------------------------
// Flash attention. All 8 waves of a block share one (b,h); wave w owns the
// 16-query block  q0 = ((blockIdx.x&7)*8 + w)*16.  Per 32-key step, K (32x128)
// and Vt (128x32) tiles are TDM-staged into double-buffered LDS once per block
// and consumed by all waves; online softmax per wave.
// Q,K: [B,H,T,D] f16;  Vt: [B,H,D,T] f16;  Yw: [B,T,C] f16.
// ---------------------------------------------------------------------------
__global__ void __launch_bounds__(256)
attn_kernel(const h16* __restrict__ Q, const h16* __restrict__ K,
            const h16* __restrict__ Vt, const int* __restrict__ mask,
            h16* __restrict__ Yw) {
    int tid  = threadIdx.x;
    int lane = tid & 31;
    int wave = tid >> 5;
    int bh = blockIdx.x >> 3;                // 0..63 (b*8+h)
    int b  = bh >> 3;
    int h  = bh & (HH - 1);
    int q0 = (((blockIdx.x & 7) << 3) + wave) << 4;

    const h16* Qh = Q  + (size_t)bh * TT * DD;
    const h16* Kh = K  + (size_t)bh * TT * DD;
    const h16* Vh = Vt + (size_t)bh * DD * TT;
    const int* mb = mask + b * TT;

    int halfid = lane >> 4;
    int nl = lane & 15;

    __shared__ __align__(16) h16 sK[2][32][LP128];   // keys x d
    __shared__ __align__(16) h16 sV[2][128][LP32];   // d x keys
    __shared__ __align__(16) h16 plds[8][16][32];

    v16h qf[4];
#pragma unroll
    for (int dk = 0; dk < 4; ++dk)
        qf[dk] = load_a16x32(Qh, DD, q0, dk * 32, lane);

    v8f yacc[8] = {};
    float mrun[8], lrun[8];
#pragma unroll
    for (int r = 0; r < 8; ++r) { mrun[r] = -3.0e38f; lrun[r] = 0.0f; }

#if USE_TDM
    i32x8 g1K = tdm_make_g1(DD, TT, DD, 128, 32, 5, 3);  // 256B rows, pad 16B
    i32x8 g1V = tdm_make_g1(TT, DD, TT, 32, 128, 3, 3);  // 64B rows,  pad 16B
    auto stage = [&](int buf, int kb) {
        if (wave == 0) {
            tdm_issue(g1K, (unsigned)(uintptr_t)&sK[buf][0][0],
                      Kh + (size_t)kb * DD);
            tdm_issue(g1V, (unsigned)(uintptr_t)&sV[buf][0][0],
                      Vh + kb);
        }
    };
#else
    auto stage = [&](int buf, int kb) {
#pragma unroll
        for (int i = 0; i < 2; ++i) {
            int c = tid * 2 + i;                 // K: 512 chunks of 16B
            int row = c >> 4, col = (c & 15) * 8;
            v8h v = *(const v8h*)(Kh + (size_t)(kb + row) * DD + col);
            *(v8h*)&sK[buf][row][col] = v;
        }
#pragma unroll
        for (int i = 0; i < 2; ++i) {
            int c = tid * 2 + i;                 // V: 512 chunks of 16B
            int row = c >> 2, col = (c & 3) * 8;
            v8h v = *(const v8h*)(Vh + (size_t)row * TT + kb + col);
            *(v8h*)&sV[buf][row][col] = v;
        }
    };
#endif

    stage(0, 0);
#if USE_TDM
    if (wave == 0) __builtin_amdgcn_s_wait_tensorcnt(0);
#endif
    __syncthreads();

    const float sc = 0.08838834764831845f;          // 1/sqrt(128)
    int buf = 0;

    for (int kb = 0; kb < TT; kb += 32) {
        if (kb + 32 < TT) stage(buf ^ 1, kb + 32);

        // ---- S = Q K^T (two 16-key column tiles) from LDS ----
        v8f s0 = {}, s1 = {};
#pragma unroll
        for (int dk = 0; dk < 4; ++dk) {
            v16h k0f = load_b32x16(&sK[buf][0][0], LP128, 0,  dk * 32, lane);
            v16h k1f = load_b32x16(&sK[buf][0][0], LP128, 16, dk * 32, lane);
            s0 = wmma_f16(qf[dk], k0f, s0);
            s1 = wmma_f16(qf[dk], k1f, s1);
        }

        float mk0 = (mb[kb + nl]      == 0) ? -3.0e38f : 0.0f;
        float mk1 = (mb[kb + 16 + nl] == 0) ? -3.0e38f : 0.0f;

        float p0[8], p1[8], corr[8];
#pragma unroll
        for (int r = 0; r < 8; ++r) {
            float a0 = s0[r] * sc + mk0;
            float a1 = s1[r] * sc + mk1;
            float mx = fmaxf(a0, a1);
#pragma unroll
            for (int off = 1; off < 16; off <<= 1)
                mx = fmaxf(mx, __shfl_xor(mx, off, 32));
            float mnew = fmaxf(mrun[r], mx);
            float c  = __expf(mrun[r] - mnew);
            float e0 = __expf(a0 - mnew);
            float e1 = __expf(a1 - mnew);
            float rs = e0 + e1;
#pragma unroll
            for (int off = 1; off < 16; off <<= 1)
                rs += __shfl_xor(rs, off, 32);
            lrun[r] = lrun[r] * c + rs;
            mrun[r] = mnew;
            corr[r] = c;
            p0[r] = e0;
            p1[r] = e1;
        }
#pragma unroll
        for (int j = 0; j < 8; ++j)
#pragma unroll
            for (int r = 0; r < 8; ++r)
                yacc[j][r] *= corr[r];

        // ---- transpose P (C-layout -> A-frag layout) via per-wave LDS ----
#pragma unroll
        for (int r = 0; r < 8; ++r) {
            plds[wave][8 * halfid + r][nl]      = (h16)p0[r];
            plds[wave][8 * halfid + r][16 + nl] = (h16)p1[r];
        }
        v16h pf = load_a16x32(&plds[wave][0][0], 32, 0, 0, lane);

        // ---- Y += P * V from LDS ----
#pragma unroll
        for (int j = 0; j < 8; ++j) {
            v16h vf = load_b32x16(&sV[buf][0][0], LP32, j * 16, 0, lane);
            yacc[j] = wmma_f16(pf, vf, yacc[j]);
        }

#if USE_TDM
        if (wave == 0) __builtin_amdgcn_s_wait_tensorcnt(0);
#endif
        __syncthreads();
        buf ^= 1;
    }

#pragma unroll
    for (int j = 0; j < 8; ++j) {
        int d = j * 16 + nl;
#pragma unroll
        for (int r = 0; r < 8; ++r) {
            int t = q0 + 8 * halfid + r;
            float y = yacc[j][r] / lrun[r];
            Yw[((size_t)(b * TT + t)) * CC + h * DD + d] = (h16)y;
        }
    }
}

// ---------------------------------------------------------------------------
// Launch
// ---------------------------------------------------------------------------
extern "C" void kernel_launch(void* const* d_in, const int* in_sizes, int n_in,
                              void* d_out, int out_size, void* d_ws, size_t ws_size,
                              hipStream_t stream) {
    (void)in_sizes; (void)n_in; (void)out_size; (void)ws_size;

    const float* x    = (const float*)d_in[0];
    const int*   mask = (const int*)d_in[1];
    const float* Wq   = (const float*)d_in[2];
    const float* bq   = (const float*)d_in[3];
    const float* Wk   = (const float*)d_in[4];
    const float* bk   = (const float*)d_in[5];
    const float* Wv   = (const float*)d_in[6];
    const float* bv   = (const float*)d_in[7];
    const float* Wp   = (const float*)d_in[8];
    const float* bp   = (const float*)d_in[9];

    const size_t NX = (size_t)MM * CC;   // 8M elems
    const size_t NW = (size_t)CC * CC;   // 1M elems

    h16* ws  = (h16*)d_ws;
    h16* xh  = ws;                 // [MM][CC]
    h16* wqh = xh  + NX;           // [CC][CC]
    h16* wkh = wqh + NW;
    h16* wvh = wkh + NW;
    h16* wph = wvh + NW;
    h16* Qs  = wph + NW;           // [B,H,T,D]
    h16* Ks  = Qs  + NX;           // [B,H,T,D]
    h16* Vts = Ks  + NX;           // [B,H,D,T]
    h16* Yw  = Vts + NX;           // [B,T,C]

    cast_f32_f16_kernel<<<dim3(NX / 1024), dim3(256), 0, stream>>>(x, xh, (int)NX);
    cast_f32_f16_kernel<<<dim3(NW / 1024), dim3(256), 0, stream>>>(Wq, wqh, (int)NW);
    cast_f32_f16_kernel<<<dim3(NW / 1024), dim3(256), 0, stream>>>(Wk, wkh, (int)NW);
    cast_f32_f16_kernel<<<dim3(NW / 1024), dim3(256), 0, stream>>>(Wv, wvh, (int)NW);
    cast_f32_f16_kernel<<<dim3(NW / 1024), dim3(256), 0, stream>>>(Wp, wph, (int)NW);

    dim3 ggrid(CC / 128, MM / 128);   // (8, 64)
    dim3 gblk(256);

    gemm_kernel<<<ggrid, gblk, 0, stream>>>(xh, wqh, bq, (void*)Qs, 0);
    gemm_kernel<<<ggrid, gblk, 0, stream>>>(xh, wkh, bk, (void*)Ks, 0);
    gemm_kernel<<<ggrid, gblk, 0, stream>>>(xh, wvh, bv, (void*)Vts, 1);

    attn_kernel<<<dim3(512), dim3(256), 0, stream>>>(Qs, Ks, Vts, mask, Yw);

    gemm_kernel<<<ggrid, gblk, 0, stream>>>(Yw, wph, bp, d_out, 2);
}
